// Transformer_47708496724396
// MI455X (gfx1250) — compile-verified, matrix operand order checked by reading, not run on hardware
//
#include <hip/hip_runtime.h>
#include <hip/hip_bf16.h>

// ---------------------------------------------------------------------------
// Types for CDNA5 WMMA (wave32): v_wmma_f32_16x16x32_bf16
// ---------------------------------------------------------------------------
typedef __attribute__((ext_vector_type(16))) __bf16 v16bf;
typedef __attribute__((ext_vector_type(8)))  float  v8f;

// Model constants
#define DMODEL 512
#define NHEAD  8
#define DK     64
#define NLAYER 6
#define FFDIM  2048
#define BATCH  2
#define SEQ    1024
#define VOCAB  32000
#define MTOK   (BATCH * SEQ)   // 2048 tokens

// float -> bf16 (round to nearest even), raw ushort bit pattern.
__device__ __forceinline__ unsigned short f2bf(float f) {
    unsigned int u = __float_as_uint(f);
    u += 0x7FFFu + ((u >> 16) & 1u);
    return (unsigned short)(u >> 16);
}
// Pack two floats into one DWORD of bf16 pairs (lo in [15:0]).
__device__ __forceinline__ unsigned int f2bf2(float lo, float hi) {
    return (unsigned int)f2bf(lo) | ((unsigned int)f2bf(hi) << 16);
}

// Load one 16x32 bf16 A-style fragment from an LDS tile.
// ISA layout (CDNA5 ISA): lane r(0..15)/r+16 holds row M=r; K chunks are
// [half*8 .. half*8+7] and [16+half*8 .. 16+half*8+7] -> two 16B LDS reads.
__device__ __forceinline__ v16bf load_frag(const unsigned short* lds,
                                           int rowBase, int strideU,
                                           int kBase, int r, int half) {
    const unsigned short* p = lds + (size_t)(rowBase + r) * strideU + kBase;
    union { v16bf v; uint4 q[2]; } f;
    f.q[0] = *(const uint4*)(p + half * 8);
    f.q[1] = *(const uint4*)(p + 16 + half * 8);
    return f.v;
}

// ---------------------------------------------------------------------------
// Embedding gather + positional encoding:  out[row,:] = emb[tok[row],:] + pe[s,:]
// ---------------------------------------------------------------------------
__global__ __launch_bounds__(256) void embed_pe(const int* __restrict__ toks,
                                                const float* __restrict__ emb,
                                                const float* __restrict__ pe,
                                                float* __restrict__ out) {
    const int row = blockIdx.x;           // b*SEQ + s
    const int s   = row & (SEQ - 1);
    const int tok = toks[row];
    const float* e = emb + (size_t)tok * DMODEL;
    const float* p = pe + (size_t)s * DMODEL;
    float* o = out + (size_t)row * DMODEL;
    for (int i = threadIdx.x; i < DMODEL; i += 256)
        o[i] = e[i] + p[i];
}

// ---------------------------------------------------------------------------
// Tiled WMMA GEMM:  C[M,N] = act(A[M,K] * W[N,K]^T + bias[N])
// Block = 256 threads (8 waves), tile 128x128x64. Wave grid 4(m) x 2(n),
// each wave owns a 32x64 sub-tile = 2x4 accumulators; 16 WMMA per stage.
// fp32 global -> bf16 LDS conversion on the fly (packed b32 stores);
// fp32 accumulate; next-K-tile prefetch (global_prefetch_b8) across barrier.
// M % 128 == N % 128 == K % 64 == 0 (true for every GEMM in this model).
// ---------------------------------------------------------------------------
template <bool RELU>
__global__ __launch_bounds__(256) void gemm_nt(const float* __restrict__ A,
                                               const float* __restrict__ W,
                                               const float* __restrict__ bias,
                                               float* __restrict__ C,
                                               int M, int N, int K) {
    __shared__ __align__(16) unsigned short As[128 * 64];
    __shared__ __align__(16) unsigned short Bs[128 * 64];

    const int tid  = threadIdx.x;
    const int lane = tid & 31;
    const int wid  = tid >> 5;
    const int half = (lane >> 4) & 1;
    const int r    = lane & 15;
    const int m0   = blockIdx.y * 128;
    const int n0   = blockIdx.x * 128;
    const int wm   = wid & 3;   // 0..3 -> 32-row strip
    const int wn   = wid >> 2;  // 0..1 -> 64-col strip

    v8f zero = {};
    v8f acc[2][4];
#pragma unroll
    for (int mi = 0; mi < 2; ++mi)
#pragma unroll
        for (int ni = 0; ni < 4; ++ni) acc[mi][ni] = zero;

    for (int k0 = 0; k0 < K; k0 += 64) {
        const bool pf_next = (k0 + 64) < K;
        // Stage A tile (128x64 fp32 -> bf16). 2048 float4, 8 per thread.
#pragma unroll
        for (int i = 0; i < 8; ++i) {
            int qq = tid + i * 256;
            int row = qq >> 4, c4 = (qq & 15) * 4;
            const float* ga = A + (size_t)(m0 + row) * K + k0 + c4;
            float4 f = *(const float4*)ga;
            unsigned int* d = (unsigned int*)(As + row * 64 + c4);
            d[0] = f2bf2(f.x, f.y);
            d[1] = f2bf2(f.z, f.w);
            if (pf_next) __builtin_prefetch(ga + 64, 0, 3);
        }
        // Stage W tile (rows of W give B = W^T in WMMA B-layout).
#pragma unroll
        for (int i = 0; i < 8; ++i) {
            int qq = tid + i * 256;
            int row = qq >> 4, c4 = (qq & 15) * 4;
            const float* gw = W + (size_t)(n0 + row) * K + k0 + c4;
            float4 f = *(const float4*)gw;
            unsigned int* d = (unsigned int*)(Bs + row * 64 + c4);
            d[0] = f2bf2(f.x, f.y);
            d[1] = f2bf2(f.z, f.w);
            if (pf_next) __builtin_prefetch(gw + 64, 0, 3);
        }
        __syncthreads();

#pragma unroll
        for (int ks = 0; ks < 2; ++ks) {
            v16bf af[2], bf[4];
#pragma unroll
            for (int mi = 0; mi < 2; ++mi)
                af[mi] = load_frag(As, wm * 32 + mi * 16, 64, ks * 32, r, half);
#pragma unroll
            for (int ni = 0; ni < 4; ++ni)
                bf[ni] = load_frag(Bs, wn * 64 + ni * 16, 64, ks * 32, r, half);

#pragma unroll
            for (int mi = 0; mi < 2; ++mi)
#pragma unroll
                for (int ni = 0; ni < 4; ++ni)
                    acc[mi][ni] = __builtin_amdgcn_wmma_f32_16x16x32_bf16(
                        false, af[mi], false, bf[ni], (short)0, acc[mi][ni],
                        false, false);
        }
        __syncthreads();
    }

    // Epilogue: bias (+ReLU), coalesced stores along N.
#pragma unroll
    for (int mi = 0; mi < 2; ++mi) {
#pragma unroll
        for (int ni = 0; ni < 4; ++ni) {
            int col = n0 + wn * 64 + ni * 16 + r;
            float b = bias ? bias[col] : 0.0f;
#pragma unroll
            for (int g = 0; g < 8; ++g) {
                int row = m0 + wm * 32 + mi * 16 + g + 8 * half;
                float v = acc[mi][ni][g] + b;
                if (RELU) v = fmaxf(v, 0.0f);
                C[(size_t)row * N + col] = v;
            }
        }
    }
}

// ---------------------------------------------------------------------------
// Flash attention (one kernel, no score materialization).
// Block = 128 threads (4 waves); block handles (b, h, 64 q-rows); each wave
// owns 16 q-rows. Streams 64-key blocks: S = Q*K^T via WMMA, online softmax
// on the C-layout fragments (shfl-xor reductions within 16-lane groups),
// P written bf16 to a wave-private LDS tile, context via WMMA P*Vt.
// Scale = 1/sqrt(DK) = 0.125. Layout: Q/K/V/O are [B, S, D] with head slice
// cols [h*64, h*64+64) -> output lands pre-concatenated across heads.
// ---------------------------------------------------------------------------
__global__ __launch_bounds__(128) void flash_attn(const float* __restrict__ Q,
                                                  const float* __restrict__ K,
                                                  const float* __restrict__ V,
                                                  float* __restrict__ O,
                                                  int Sq, int Sk, int causal) {
    __shared__ __align__(16) unsigned short Qs[64 * 64];
    __shared__ __align__(16) unsigned short Ks[64 * 64];
    __shared__ __align__(16) unsigned short Vts[64 * 64];   // transposed: [dk][key]
    __shared__ __align__(16) unsigned short Ps[4][16 * 64]; // per-wave P tile

    const int tid  = threadIdx.x;
    const int lane = tid & 31;
    const int wid  = tid >> 5;
    const int half = (lane >> 4) & 1;
    const int r    = lane & 15;
    const int qb   = blockIdx.x, h = blockIdx.y, b = blockIdx.z;
    const int q0   = qb * 64;
    const int qrow0 = wid * 16;

    const size_t baseQ  = ((size_t)b * Sq + q0) * DMODEL + h * DK;
    const size_t baseKV = (size_t)b * Sk * DMODEL + h * DK;

    // Stage Q (64x64): 1024 float4, 8 per thread; packed b32 LDS stores.
#pragma unroll
    for (int i = 0; i < 8; ++i) {
        int qq = tid + i * 128;
        int row = qq >> 4, c4 = (qq & 15) * 4;
        float4 f = *(const float4*)(Q + baseQ + (size_t)row * DMODEL + c4);
        unsigned int* d = (unsigned int*)(Qs + row * 64 + c4);
        d[0] = f2bf2(f.x, f.y);
        d[1] = f2bf2(f.z, f.w);
    }

    v8f zero = {};
    v8f o[4];
#pragma unroll
    for (int i = 0; i < 4; ++i) o[i] = zero;
    float m[8], l[8];
#pragma unroll
    for (int g = 0; g < 8; ++g) { m[g] = -1e30f; l[g] = 0.0f; }

    const int kbEnd = causal ? (qb + 1) : (Sk >> 6);
    for (int kb = 0; kb < kbEnd; ++kb) {
        __syncthreads();  // previous iteration's reads done before restage
        // Stage K (row-major, packed) and V transposed (Vt[dk][key], b16).
#pragma unroll
        for (int i = 0; i < 8; ++i) {
            int qq = tid + i * 128;
            int row = qq >> 4, c4 = (qq & 15) * 4;
            size_t goff = baseKV + (size_t)(kb * 64 + row) * DMODEL + c4;
            float4 f = *(const float4*)(K + goff);
            unsigned int* d = (unsigned int*)(Ks + row * 64 + c4);
            d[0] = f2bf2(f.x, f.y);
            d[1] = f2bf2(f.z, f.w);
            float4 g4 = *(const float4*)(V + goff);
            Vts[(c4 + 0) * 64 + row] = f2bf(g4.x);
            Vts[(c4 + 1) * 64 + row] = f2bf(g4.y);
            Vts[(c4 + 2) * 64 + row] = f2bf(g4.z);
            Vts[(c4 + 3) * 64 + row] = f2bf(g4.w);
        }
        __syncthreads();

        // S = Q * K^T for this wave's 16 q-rows x 64 keys (dk=64 -> 2 k-steps).
        v16bf aq0 = load_frag(Qs, qrow0, 64, 0, r, half);
        v16bf aq1 = load_frag(Qs, qrow0, 64, 32, r, half);
        v8f s[4];
#pragma unroll
        for (int nf = 0; nf < 4; ++nf) {
            v16bf k0f = load_frag(Ks, nf * 16, 64, 0, r, half);
            v16bf k1f = load_frag(Ks, nf * 16, 64, 32, r, half);
            v8f sc = __builtin_amdgcn_wmma_f32_16x16x32_bf16(
                false, aq0, false, k0f, (short)0, zero, false, false);
            sc = __builtin_amdgcn_wmma_f32_16x16x32_bf16(
                false, aq1, false, k1f, (short)0, sc, false, false);
            s[nf] = sc;
        }

        // Scale + causal mask + row max (C layout: row = g+8*half, col = nf*16+r).
        float rowm[8];
#pragma unroll
        for (int g = 0; g < 8; ++g) {
            float mx = -1e30f;
#pragma unroll
            for (int nf = 0; nf < 4; ++nf) {
                float v = s[nf][g] * 0.125f;
                if (causal) {
                    int kidx = kb * 64 + nf * 16 + r;
                    int qidx = q0 + qrow0 + g + 8 * half;
                    if (kidx > qidx) v = -1e30f;
                }
                s[nf][g] = v;
                mx = fmaxf(mx, v);
            }
#pragma unroll
            for (int off = 8; off >= 1; off >>= 1)
                mx = fmaxf(mx, __shfl_xor(mx, off, 32));
            rowm[g] = mx;
        }

        // Online softmax update + rescale running output.
#pragma unroll
        for (int g = 0; g < 8; ++g) {
            float nm = fmaxf(m[g], rowm[g]);
            float alpha = __expf(m[g] - nm);
            float rs = 0.0f;
#pragma unroll
            for (int nf = 0; nf < 4; ++nf) {
                float p = __expf(s[nf][g] - nm);
                s[nf][g] = p;
                rs += p;
            }
#pragma unroll
            for (int off = 8; off >= 1; off >>= 1)
                rs += __shfl_xor(rs, off, 32);
            l[g] = l[g] * alpha + rs;
            m[g] = nm;
#pragma unroll
            for (int nf = 0; nf < 4; ++nf) o[nf][g] *= alpha;
        }

        // P (C layout) -> wave-private LDS tile, bf16. Same-wave LDS ops are
        // in-order; no block barrier needed.
        unsigned short* P = &Ps[wid][0];
#pragma unroll
        for (int g = 0; g < 8; ++g) {
            int prow = g + 8 * half;
#pragma unroll
            for (int nf = 0; nf < 4; ++nf)
                P[prow * 64 + nf * 16 + r] = f2bf(s[nf][g]);
        }

        // O += P * V  (keys=64 -> 2 k-steps; Vt rows are dk).
#pragma unroll
        for (int ks = 0; ks < 2; ++ks) {
            v16bf pf = load_frag(P, 0, 64, ks * 32, r, half);
#pragma unroll
            for (int nf = 0; nf < 4; ++nf) {
                v16bf vf = load_frag(Vts, nf * 16, 64, ks * 32, r, half);
                o[nf] = __builtin_amdgcn_wmma_f32_16x16x32_bf16(
                    false, pf, false, vf, (short)0, o[nf], false, false);
            }
        }
    }

    // Normalize and store context at [b, q, h*64 + d] (heads pre-concatenated).
#pragma unroll
    for (int nf = 0; nf < 4; ++nf) {
#pragma unroll
        for (int g = 0; g < 8; ++g) {
            int row = q0 + qrow0 + g + 8 * half;
            O[((size_t)b * Sq + row) * DMODEL + h * DK + nf * 16 + r] =
                o[nf][g] / l[g];
        }
    }
}

// ---------------------------------------------------------------------------
// Fused residual-add + LayerNorm:  out = LN(x + res) * g + b   (D = 512)
// One block per token row; safe for out == x (reads before barrier+writes).
// ---------------------------------------------------------------------------
__global__ __launch_bounds__(256) void ln_res(const float* __restrict__ X,
                                              const float* __restrict__ R,
                                              const float* __restrict__ gamma,
                                              const float* __restrict__ beta,
                                              float* __restrict__ out) {
    __shared__ float s1[256];
    __shared__ float s2[256];
    const int row = blockIdx.x, tid = threadIdx.x;
    const float* x = X + (size_t)row * DMODEL;
    const float* rr = R + (size_t)row * DMODEL;
    float t0 = x[tid] + rr[tid];
    float t1 = x[tid + 256] + rr[tid + 256];
    s1[tid] = t0 + t1;
    s2[tid] = t0 * t0 + t1 * t1;
    __syncthreads();
    for (int st = 128; st > 0; st >>= 1) {
        if (tid < st) { s1[tid] += s1[tid + st]; s2[tid] += s2[tid + st]; }
        __syncthreads();
    }
    float mean = s1[0] * (1.0f / DMODEL);
    float var  = s2[0] * (1.0f / DMODEL) - mean * mean;
    float inv  = rsqrtf(var + 1e-5f);
    float* o = out + (size_t)row * DMODEL;
    o[tid]       = (t0 - mean) * inv * gamma[tid] + beta[tid];
    o[tid + 256] = (t1 - mean) * inv * gamma[tid + 256] + beta[tid + 256];
}

// ---------------------------------------------------------------------------
// Driver. Input order (setup_inputs dict order, nested dicts flattened in
// insertion order):
//  0 src  1 tgt  2 tgt_mask  3 embedding  4 pe
//  enc (5..20):  Wq bq Wk bk Wv bv Wo bo  W1 b1 W2 b2  ln1_g ln1_b ln2_g ln2_b
//  dec (21..46): s_{Wq..bo} c_{Wq..bo} W1 b1 W2 b2 ln1_g ln1_b ln2_g ln2_b ln3_g ln3_b
//  47 out_W  48 out_b
// ---------------------------------------------------------------------------
extern "C" void kernel_launch(void* const* d_in, const int* in_sizes, int n_in,
                              void* d_out, int out_size, void* d_ws, size_t ws_size,
                              hipStream_t stream) {
    (void)in_sizes; (void)n_in; (void)out_size; (void)ws_size;

    const int* src = (const int*)d_in[0];
    const int* tgt = (const int*)d_in[1];
    const float* emb = (const float*)d_in[3];
    const float* pe  = (const float*)d_in[4];
    auto P = [&](int i) { return (const float*)d_in[i]; };

    const size_t DD = (size_t)DMODEL * DMODEL;   // 512*512
    const size_t FD = (size_t)FFDIM * DMODEL;    // 2048*512
    const size_t MD = (size_t)MTOK * DMODEL;     // token activations

    float* ws  = (float*)d_ws;
    float* x   = ws;            // current activations [2048, 512]
    float* mem = ws + 1 * MD;   // encoder output
    float* q   = ws + 2 * MD;
    float* k   = ws + 3 * MD;
    float* v   = ws + 4 * MD;
    float* ctx = ws + 5 * MD;
    float* ao  = ws + 6 * MD;   // sublayer output before residual+LN
    float* h1  = ws + 7 * MD;   // FFN hidden [2048, 2048] (4*MD)

    auto G = [](int M, int N) { return dim3(N / 128, M / 128); };
    const dim3 attG(SEQ / 64, NHEAD, BATCH);

    // ---- Encoder ----
    embed_pe<<<MTOK, 256, 0, stream>>>(src, emb, pe, x);
    for (int l = 0; l < NLAYER; ++l) {
        const float *Wq = P(5)  + l * DD, *bq = P(6)  + l * DMODEL;
        const float *Wk = P(7)  + l * DD, *bk = P(8)  + l * DMODEL;
        const float *Wv = P(9)  + l * DD, *bv = P(10) + l * DMODEL;
        const float *Wo = P(11) + l * DD, *bo = P(12) + l * DMODEL;
        const float *W1 = P(13) + l * FD, *b1 = P(14) + l * FFDIM;
        const float *W2 = P(15) + l * FD, *b2 = P(16) + l * DMODEL;

        gemm_nt<false><<<G(MTOK, DMODEL), 256, 0, stream>>>(x, Wq, bq, q, MTOK, DMODEL, DMODEL);
        gemm_nt<false><<<G(MTOK, DMODEL), 256, 0, stream>>>(x, Wk, bk, k, MTOK, DMODEL, DMODEL);
        gemm_nt<false><<<G(MTOK, DMODEL), 256, 0, stream>>>(x, Wv, bv, v, MTOK, DMODEL, DMODEL);
        flash_attn<<<attG, 128, 0, stream>>>(q, k, v, ctx, SEQ, SEQ, 0);
        gemm_nt<false><<<G(MTOK, DMODEL), 256, 0, stream>>>(ctx, Wo, bo, ao, MTOK, DMODEL, DMODEL);
        ln_res<<<MTOK, 256, 0, stream>>>(x, ao, P(17) + l * DMODEL, P(18) + l * DMODEL, x);
        gemm_nt<true><<<G(MTOK, FFDIM), 256, 0, stream>>>(x, W1, b1, h1, MTOK, FFDIM, DMODEL);
        gemm_nt<false><<<G(MTOK, DMODEL), 256, 0, stream>>>(h1, W2, b2, ao, MTOK, DMODEL, FFDIM);
        ln_res<<<MTOK, 256, 0, stream>>>(x, ao, P(19) + l * DMODEL, P(20) + l * DMODEL, x);
    }
    hipMemcpyAsync(mem, x, MD * sizeof(float), hipMemcpyDeviceToDevice, stream);

    // ---- Decoder ----
    embed_pe<<<MTOK, 256, 0, stream>>>(tgt, emb, pe, x);
    for (int l = 0; l < NLAYER; ++l) {
        const float *sWq = P(21) + l * DD, *sbq = P(22) + l * DMODEL;
        const float *sWk = P(23) + l * DD, *sbk = P(24) + l * DMODEL;
        const float *sWv = P(25) + l * DD, *sbv = P(26) + l * DMODEL;
        const float *sWo = P(27) + l * DD, *sbo = P(28) + l * DMODEL;
        const float *cWq = P(29) + l * DD, *cbq = P(30) + l * DMODEL;
        const float *cWk = P(31) + l * DD, *cbk = P(32) + l * DMODEL;
        const float *cWv = P(33) + l * DD, *cbv = P(34) + l * DMODEL;
        const float *cWo = P(35) + l * DD, *cbo = P(36) + l * DMODEL;
        const float *W1 = P(37) + l * FD, *b1 = P(38) + l * FFDIM;
        const float *W2 = P(39) + l * FD, *b2 = P(40) + l * DMODEL;

        // masked self-attention (causal == tgt_mask tril)
        gemm_nt<false><<<G(MTOK, DMODEL), 256, 0, stream>>>(x, sWq, sbq, q, MTOK, DMODEL, DMODEL);
        gemm_nt<false><<<G(MTOK, DMODEL), 256, 0, stream>>>(x, sWk, sbk, k, MTOK, DMODEL, DMODEL);
        gemm_nt<false><<<G(MTOK, DMODEL), 256, 0, stream>>>(x, sWv, sbv, v, MTOK, DMODEL, DMODEL);
        flash_attn<<<attG, 128, 0, stream>>>(q, k, v, ctx, SEQ, SEQ, 1);
        gemm_nt<false><<<G(MTOK, DMODEL), 256, 0, stream>>>(ctx, sWo, sbo, ao, MTOK, DMODEL, DMODEL);
        ln_res<<<MTOK, 256, 0, stream>>>(x, ao, P(41) + l * DMODEL, P(42) + l * DMODEL, x);

        // cross-attention against encoder memory
        gemm_nt<false><<<G(MTOK, DMODEL), 256, 0, stream>>>(x,   cWq, cbq, q, MTOK, DMODEL, DMODEL);
        gemm_nt<false><<<G(MTOK, DMODEL), 256, 0, stream>>>(mem, cWk, cbk, k, MTOK, DMODEL, DMODEL);
        gemm_nt<false><<<G(MTOK, DMODEL), 256, 0, stream>>>(mem, cWv, cbv, v, MTOK, DMODEL, DMODEL);
        flash_attn<<<attG, 128, 0, stream>>>(q, k, v, ctx, SEQ, SEQ, 0);
        gemm_nt<false><<<G(MTOK, DMODEL), 256, 0, stream>>>(ctx, cWo, cbo, ao, MTOK, DMODEL, DMODEL);
        ln_res<<<MTOK, 256, 0, stream>>>(x, ao, P(43) + l * DMODEL, P(44) + l * DMODEL, x);

        // FFN
        gemm_nt<true><<<G(MTOK, FFDIM), 256, 0, stream>>>(x, W1, b1, h1, MTOK, FFDIM, DMODEL);
        gemm_nt<false><<<G(MTOK, DMODEL), 256, 0, stream>>>(h1, W2, b2, ao, MTOK, DMODEL, FFDIM);
        ln_res<<<MTOK, 256, 0, stream>>>(x, ao, P(45) + l * DMODEL, P(46) + l * DMODEL, x);
    }

    // ---- Vocab projection: logits = x @ out_W^T + out_b ----
    gemm_nt<false><<<G(MTOK, VOCAB), 256, 0, stream>>>(
        x, P(47), P(48), (float*)d_out, MTOK, VOCAB, DMODEL);
}
